// SpatialTemporalEncoderLayer_10788957848223
// MI455X (gfx1250) — matrix operand order repaired and checked by reading, not compiled, not run on hardware
//
#include <hip/hip_runtime.h>
#include <hip/hip_bf16.h>
#include <stdint.h>

#define NHEADS 4
#define HDIM   12
#define NEXP   16
#define CAPN   4
#define THRESH 0.2f

typedef __attribute__((ext_vector_type(16))) __bf16        v16bf;
typedef __attribute__((ext_vector_type(8)))  float         v8f;
typedef __attribute__((ext_vector_type(8)))  unsigned int  v8u;

__device__ __forceinline__ unsigned short f32_to_bf16(float f) {
  unsigned int u = __float_as_uint(f);
  unsigned int r = u + 0x7FFFu + ((u >> 16) & 1u);
  return (unsigned short)(r >> 16);
}

__device__ __forceinline__ float wave_sum32(float v) {
  #pragma unroll
  for (int off = 16; off; off >>= 1) v += __shfl_xor(v, off, 32);
  return v;
}

__device__ __forceinline__ void store_val(float* p, float v)          { *p = v; }
__device__ __forceinline__ void store_val(unsigned short* p, float v) { *p = f32_to_bf16(v); }

// ---------------------------------------------------------------------------
// pack/pad f32 -> bf16 (zero-fill pad region)
// ---------------------------------------------------------------------------
__global__ void __launch_bounds__(256)
pack_bf16_pad_kernel(const float* __restrict__ src, unsigned short* __restrict__ dst,
                     long long srows, int scols, long long drows, int dcols)
{
  long long idx = (long long)blockIdx.x * 256 + threadIdx.x;
  long long total = drows * (long long)dcols;
  if (idx >= total) return;
  long long r = idx / dcols;
  int c = (int)(idx - r * dcols);
  float v = (r < srows && c < scols) ? src[r * (long long)scols + c] : 0.f;
  dst[idx] = f32_to_bf16(v);
}

// ---------------------------------------------------------------------------
// WMMA GEMM: C[M,N] = A[M,K]bf16 @ W[K,N]bf16 (+bias +resid), f32 accumulate.
// Requirements: M % 128 == 0, K % 64 == 0, Ns (W stride) % 128 == 0,
// grid.x*128 <= Ns. Only epilogue guards gn < N.
// A tile staged via CDNA5 async global->LDS copy (ASYNCcnt), B tile staged
// with 16B loads + LDS transpose stores. Double-buffered, K-chunk 64.
// ---------------------------------------------------------------------------
#define BMT 128
#define BNT 128
#define BKC 64

__global__ void __launch_bounds__(256)
gemm_bf16_wmma(const unsigned short* __restrict__ A, const unsigned short* __restrict__ W,
               const float* __restrict__ bias, const float* __restrict__ resid,
               float* __restrict__ C,
               int M, int N, int Ns, int K,
               long long sA, long long sB, long long sBias, long long sC)
{
  const int ez = blockIdx.z;
  A += (long long)ez * sA;
  W += (long long)ez * sB;
  C += (long long)ez * sC;
  const float* bptr = bias  ? bias  + (long long)ez * sBias : nullptr;
  const float* rptr = resid ? resid + (long long)ez * sC    : nullptr;

  __shared__ __align__(16) unsigned short As[2][BMT * BKC]; // [row][k]   32KB
  __shared__ __align__(16) unsigned short Bs[2][BNT * BKC]; // [col][k]   32KB

  const int tid   = threadIdx.x;
  const int lane  = tid & 31;
  const int wave  = tid >> 5;
  const int waveM = wave >> 2;   // 0..1
  const int waveN = wave & 3;    // 0..3
  const int tileM0 = blockIdx.y * BMT;
  const int tileN0 = blockIdx.x * BNT;
  const int half = lane >> 4;
  const int l16  = lane & 15;

  v8f acc[4][2];
  #pragma unroll
  for (int i = 0; i < 4; ++i)
    #pragma unroll
    for (int j = 0; j < 2; ++j)
      #pragma unroll
      for (int r = 0; r < 8; ++r) acc[i][j][r] = 0.f;

  // ---- A tile: verbatim async copy global -> LDS (4 x b128 per thread) ----
  auto stageA = [&](int kc, int buf) {
    #pragma unroll
    for (int i = 0; i < 4; ++i) {
      int c  = tid + 256 * i;          // 1024 chunks of 16B
      int r  = c >> 3;                 // 8 chunks (128B) per row
      int kk = (c & 7) * 8;            // ushort offset in row
      const unsigned short* gp = A + ((long long)(tileM0 + r) * K + kc * BKC + kk);
      unsigned ldsoff = (unsigned)(uintptr_t)&As[buf][r * BKC + kk];
      unsigned long long ga = (unsigned long long)(uintptr_t)gp;
      asm volatile("global_load_async_to_lds_b128 %0, %1, off"
                   :: "v"(ldsoff), "v"(ga) : "memory");
    }
  };

  // ---- B tile: 16B loads, transpose-store to LDS [col][k] ----
  auto stageB = [&](int kc, int buf) {
    #pragma unroll
    for (int i = 0; i < 4; ++i) {
      int c    = tid + 256 * i;        // 1024 chunks of 16B
      int krow = c >> 4;               // 16 chunks (256B) per k-row
      int n0   = (c & 15) * 8;
      const unsigned short* gp = W + ((long long)(kc * BKC + krow) * Ns + tileN0 + n0);
      if (i == 0 && (kc + 1) * BKC < K)
        __builtin_prefetch(W + ((long long)((kc + 1) * BKC + krow) * Ns + tileN0 + n0), 0, 1);
      uint4 d = *(const uint4*)gp;
      union { uint4 u; unsigned short s[8]; } uu; uu.u = d;
      #pragma unroll
      for (int j = 0; j < 8; ++j) Bs[buf][(n0 + j) * BKC + krow] = uu.s[j];
    }
  };

  auto compute = [&](int cur) {
    #pragma unroll
    for (int ks = 0; ks < BKC; ks += 32) {
      v16bf bfrag[2];
      #pragma unroll
      for (int nt = 0; nt < 2; ++nt) {
        int col = waveN * 32 + nt * 16 + l16;
        const unsigned int* p = (const unsigned int*)&Bs[cur][col * BKC + ks + 16 * half];
        v8u u;
        #pragma unroll
        for (int j = 0; j < 8; ++j) u[j] = p[j];
        bfrag[nt] = __builtin_bit_cast(v16bf, u);
      }
      #pragma unroll
      for (int mt = 0; mt < 4; ++mt) {
        int row = waveM * 64 + mt * 16 + l16;
        const unsigned int* ap = (const unsigned int*)&As[cur][row * BKC + ks];
        v8u u;
        #pragma unroll
        for (int v = 0; v < 8; ++v) u[v] = ap[8 * half + v];
        v16bf afrag = __builtin_bit_cast(v16bf, u);
        #pragma unroll
        for (int nt = 0; nt < 2; ++nt)
          acc[mt][nt] = __builtin_amdgcn_wmma_f32_16x16x32_bf16(
              false, afrag, false, bfrag[nt], (short)0, acc[mt][nt], false, false);
      }
    }
  };

  const int nk = K / BKC;
  stageA(0, 0);
  stageB(0, 0);
  for (int i = 0; i < nk; ++i) {
    int cur = i & 1;
    asm volatile("s_wait_asynccnt 0" ::: "memory");
    __syncthreads();
    if (i + 1 < nk) {
      stageA(i + 1, 1 - cur);
      stageB(i + 1, 1 - cur);
    }
    compute(cur);
    __syncthreads();
  }

  // ---- epilogue (C layout: VGPR r -> M = r + 8*half, N = l16) ----
  #pragma unroll
  for (int mt = 0; mt < 4; ++mt) {
    #pragma unroll
    for (int nt = 0; nt < 2; ++nt) {
      int gn = tileN0 + waveN * 32 + nt * 16 + l16;
      if (gn >= N) continue;
      float bv = bptr ? bptr[gn] : 0.f;
      #pragma unroll
      for (int r = 0; r < 8; ++r) {
        int gm = tileM0 + waveM * 64 + mt * 16 + r + 8 * half;
        long long ci = (long long)gm * N + gn;
        float rv = rptr ? rptr[ci] : 0.f;
        C[ci] = acc[mt][nt][r] + bv + rv;
      }
    }
  }
}

// ---------------------------------------------------------------------------
// Attention: one wave per (group, head); lane = query; online softmax.
// Writes bf16 output padded to 64 cols (cols 48..63 zeroed by head 0).
// ---------------------------------------------------------------------------
__global__ void __launch_bounds__(256)
attn_softmax_kernel(const float* __restrict__ qkv, unsigned short* __restrict__ out,
                    int mode, int Bn, int Tn, int Sn)
{
  const int QW = 144, OP = 64;
  int gw   = blockIdx.x * (blockDim.x >> 5) + (threadIdx.x >> 5);
  int lane = threadIdx.x & 31;
  int ngroups = (mode == 0) ? Bn * Sn : Bn * Tn;
  if (gw >= ngroups * NHEADS) return;
  int g = gw / NHEADS, h = gw % NHEADS;

  long long base; int step, L, causal;
  if (mode == 0) { int b = g / Sn, s = g % Sn; base = (long long)b * Tn * Sn + s; step = Sn; L = Tn; causal = 1; }
  else           { base = (long long)g * Sn; step = 1; L = Sn; causal = 0; }

  int  qi = lane;
  bool active = qi < L;
  long long ti = base + (long long)qi * step;

  float q[HDIM];
  const float* qp = qkv + ti * QW + h * HDIM;
  #pragma unroll
  for (int d = 0; d < HDIM; ++d) q[d] = active ? qp[d] : 0.f;

  const float scale = rsqrtf((float)HDIM);
  float m = -3.0e38f, den = 0.f, accv[HDIM];
  #pragma unroll
  for (int d = 0; d < HDIM; ++d) accv[d] = 0.f;

  for (int tk = 0; tk < L; ++tk) {
    long long tki = base + (long long)tk * step;
    const float* kp = qkv + tki * QW + 48 + h * HDIM;
    const float* vp = qkv + tki * QW + 96 + h * HDIM;
    float s = 0.f;
    #pragma unroll
    for (int d = 0; d < HDIM; ++d) s += q[d] * kp[d];
    s *= scale;
    bool valid = active && (!causal || tk <= qi);
    if (valid) {
      float mn   = fmaxf(m, s);
      float corr = __expf(m - mn);
      float p    = __expf(s - mn);
      den = den * corr + p;
      #pragma unroll
      for (int d = 0; d < HDIM; ++d) accv[d] = accv[d] * corr + p * vp[d];
      m = mn;
    }
  }
  if (active) {
    float inv = 1.f / den;
    unsigned short* op = out + ti * OP + h * HDIM;
    #pragma unroll
    for (int d = 0; d < HDIM; ++d) op[d] = f32_to_bf16(accv[d] * inv);
    if (h == 0) {
      #pragma unroll
      for (int j = 0; j < 16; ++j) out[ti * OP + 48 + j] = 0;
    }
  }
}

// out = LayerNorm(a + b) over last dim 48; one wave per row.
__global__ void __launch_bounds__(256)
add_layernorm48(const float* __restrict__ a, const float* __restrict__ b,
                const float* __restrict__ g, const float* __restrict__ bb,
                float* __restrict__ out, long long ntok)
{
  long long row = (long long)blockIdx.x * (blockDim.x >> 5) + (threadIdx.x >> 5);
  int lane = threadIdx.x & 31;
  if (row >= ntok) return;
  const float* pa = a + row * 48;
  const float* pb = b + row * 48;
  bool has1 = lane < 16;
  float x0 = pa[lane] + pb[lane];
  float x1 = has1 ? pa[lane + 32] + pb[lane + 32] : 0.f;
  float mu = wave_sum32(x0 + x1) * (1.f / 48.f);
  float d0 = x0 - mu, d1 = has1 ? x1 - mu : 0.f;
  float var = wave_sum32(d0 * d0 + d1 * d1) * (1.f / 48.f);
  float rstd = rsqrtf(var + 1e-5f);
  float* po = out + row * 48;
  po[lane] = d0 * rstd * g[lane] + bb[lane];
  if (has1) po[lane + 32] = d1 * rstd * g[lane + 32] + bb[lane + 32];
}

// st-moe RMSNorm: x / max(||x||,1e-12) * sqrt(D) * g ; one block per row.
template <typename OutT>
__global__ void __launch_bounds__(256)
rmsnorm_kernel(const float* __restrict__ x, const float* __restrict__ g,
               OutT* __restrict__ out, int D)
{
  long long row = blockIdx.x;
  const float* px = x + row * (long long)D;
  float ss = 0.f;
  for (int j = threadIdx.x; j < D; j += 256) { float v = px[j]; ss += v * v; }
  ss = wave_sum32(ss);
  __shared__ float red[8];
  int wv = threadIdx.x >> 5, lane = threadIdx.x & 31;
  if (lane == 0) red[wv] = ss;
  __syncthreads();
  if (threadIdx.x == 0) {
    float t = 0.f;
    #pragma unroll
    for (int w = 0; w < 8; ++w) t += red[w];
    float nrm = fmaxf(sqrtf(t), 1e-12f);
    red[0] = sqrtf((float)D) / nrm;
  }
  __syncthreads();
  float sc = red[0];
  OutT* po = out + row * (long long)D;
  for (int j = threadIdx.x; j < D; j += 256) store_val(&po[j], px[j] * sc * g[j]);
}

// GEGLU: out[r,j] = h[r,j] * gelu(h[r,H+j]) * mb[(r/rowsPerE)][j]  (bf16 out)
__global__ void __launch_bounds__(256)
geglu_act_kernel(const float* __restrict__ h, const float* __restrict__ mb,
                 unsigned short* __restrict__ out, long long rows, int H, int rowsPerE)
{
  long long idx = (long long)blockIdx.x * blockDim.x + threadIdx.x;
  long long total = rows * (long long)H;
  if (idx >= total) return;
  long long r = idx / H;
  int j = (int)(idx - r * H);
  const float* hp = h + r * (2LL * H);
  float a = hp[j], g = hp[H + j];
  float gg = 0.5f * g * (1.f + erff(g * 0.70710678118654752f));
  const float* mp = mb + (rowsPerE ? (r / rowsPerE) * (long long)H : 0LL);
  out[idx] = f32_to_bf16(a * gg * mp[j]);
}

__global__ void __launch_bounds__(256)
add2_kernel(const float* a, const float* b, float* o, long long n)
{
  long long i = (long long)blockIdx.x * blockDim.x + threadIdx.x;
  if (i < n) o[i] = a[i] + b[i];
}

__global__ void __launch_bounds__(256)
copy_kernel(const float* a, float* b, long long n)
{
  long long i = (long long)blockIdx.x * blockDim.x + threadIdx.x;
  if (i < n) b[i] = a[i];
}

// gate logits: [rows x D] @ [D x 16], f32; one block per row.
__global__ void __launch_bounds__(256)
gate_logits_kernel(const float* __restrict__ xg, const float* __restrict__ gw,
                   float* __restrict__ logits, int D)
{
  long long row = blockIdx.x;
  int t = threadIdx.x, e = t & 15, sub = t >> 4;
  const float* px = xg + row * (long long)D;
  float acc = 0.f;
  for (int k = sub; k < D; k += 16) acc += px[k] * gw[(long long)k * NEXP + e];
  __shared__ float red[256];
  red[t] = acc;
  __syncthreads();
  for (int s = 8; s; s >>= 1) {
    if (sub < s) red[t] += red[t + s * 16];
    __syncthreads();
  }
  if (sub == 0) logits[row * NEXP + e] = red[e];
}

__global__ void __launch_bounds__(256)
init_routing_kernel(int* slot_token, float* aux, int nslots)
{
  int i = blockIdx.x * blockDim.x + threadIdx.x;
  if (i < nslots) slot_token[i] = -1;
  if (i < 2) aux[i] = 0.f;
}

__device__ __forceinline__ float hash_uniform(unsigned a, unsigned b, unsigned c)
{
  unsigned h = a * 0x9E3779B1u ^ b * 0x85EBCA77u ^ c * 0xC2B2AE3Du;
  h ^= h >> 16; h *= 0x7FEB352Du; h ^= h >> 15; h *= 0x846CA68Bu; h ^= h >> 16;
  return (float)(h >> 8) * (1.0f / 16777216.0f);
}

// Top-2 routing w/ capacity, one wave per batch row b (n == 32 tokens = lanes).
__global__ void __launch_bounds__(32)
routing_kernel(const float* __restrict__ logits, int Bn, int n,
               int* __restrict__ route_e, int* __restrict__ route_pos,
               float* __restrict__ route_gate, int* __restrict__ slot_token,
               float* __restrict__ aux)
{
  int b = blockIdx.x, lane = threadIdx.x;
  const float* lg = logits + ((long long)b * n + lane) * NEXP;
  float l[NEXP];
  float mx = -3.0e38f;
  #pragma unroll
  for (int e = 0; e < NEXP; ++e) { l[e] = lg[e]; mx = fmaxf(mx, l[e]); }
  float sum = 0.f;
  #pragma unroll
  for (int e = 0; e < NEXP; ++e) { l[e] = __expf(l[e] - mx); sum += l[e]; }
  float lse = __logf(sum) + mx;
  float raw[NEXP];
  #pragma unroll
  for (int e = 0; e < NEXP; ++e) raw[e] = l[e] / sum;

  int i1 = 0; float g1 = raw[0];
  #pragma unroll
  for (int e = 1; e < NEXP; ++e) if (raw[e] > g1) { g1 = raw[e]; i1 = e; }
  int i2 = -1; float g2 = -1.f;
  #pragma unroll
  for (int e = 0; e < NEXP; ++e) if (e != i1 && raw[e] > g2) { g2 = raw[e]; i2 = e; }

  float denom = fmaxf(g1 + g2, 1e-9f);
  float ng1 = g1 / denom, ng2 = g2 / denom;
  bool keep2 = hash_uniform((unsigned)b, (unsigned)lane, 1u) < (ng2 / THRESH);

  unsigned lt = (1u << lane) - 1u;
  int pos1 = 0, pos2 = 0; bool kept1 = false, kept2 = false;
  for (int e = 0; e < NEXP; ++e) {
    unsigned b1 = (unsigned)__ballot(i1 == e);
    unsigned b2 = (unsigned)__ballot(i2 == e && keep2);
    int c1 = __popc(b1);
    int kept1cnt = c1 < CAPN ? c1 : CAPN;
    if (i1 == e) { int p = __popc(b1 & lt); if (p < CAPN) { pos1 = p; kept1 = true; } }
    if (i2 == e && keep2) {
      int p = __popc(b2 & lt) + kept1cnt;
      if (p < CAPN) { pos2 = p; kept2 = true; }
    }
  }

  long long tidx = (long long)b * n + lane;
  long long ks = (long long)Bn * n;
  route_e[tidx]        = i1;
  route_e[ks + tidx]   = i2;
  route_pos[tidx]      = pos1;
  route_pos[ks + tidx] = pos2;
  route_gate[tidx]      = kept1 ? ng1 : 0.f;
  route_gate[ks + tidx] = kept2 ? ng2 : 0.f;
  if (kept1) slot_token[(i1 * Bn + b) * CAPN + pos1] = lane;
  if (kept2) slot_token[(i2 * Bn + b) * CAPN + pos2] = lane;

  float balpart = 0.f;
  for (int e = 0; e < NEXP; ++e) {
    float proxy = wave_sum32(raw[e]) / (float)n;
    unsigned b1 = (unsigned)__ballot(i1 == e);
    float dens = (float)__popc(b1) / (float)n;
    balpart += proxy * dens;
  }
  float z = wave_sum32(lse * lse);
  if (lane == 0) {
    atomicAdd(&aux[0], balpart * (float)(NEXP * NEXP) / ((float)Bn * NEXP));
    atomicAdd(&aux[1], z / ((float)Bn * n));
  }
}

// ei[slot, :] = bf16(xg[b*n+tok, :]) or 0 for empty slots; one block per slot.
__global__ void __launch_bounds__(256)
dispatch_gather_kernel(const float* __restrict__ xg, const int* __restrict__ slot_token,
                       unsigned short* __restrict__ ei, int Bn, int n, int D)
{
  long long slot = blockIdx.x;
  int tok = slot_token[slot];
  unsigned short* po = ei + slot * (long long)D;
  if (tok < 0) {
    for (int j = threadIdx.x; j < D; j += 256) po[j] = 0;
    return;
  }
  int bc = (int)(slot % (long long)(Bn * CAPN));
  int b = bc / CAPN;
  const float* px = xg + ((long long)b * n + tok) * D;
  for (int j = threadIdx.x; j < D; j += 256) po[j] = f32_to_bf16(px[j]);
}

// xm[tok,:] += sum_k gate_k * eo[(e_k*Bn+b)*CAP + pos_k, :]
__global__ void __launch_bounds__(256)
combine_kernel(const float* __restrict__ eo, const int* __restrict__ route_e,
               const int* __restrict__ route_pos, const float* __restrict__ route_gate,
               float* __restrict__ xm, int Bn, int n, int D)
{
  long long tok = blockIdx.x;
  int b = (int)(tok / n);
  long long kstride = (long long)Bn * n;
  float gA = route_gate[tok],            gB = route_gate[kstride + tok];
  int   eA = route_e[tok],               eB = route_e[kstride + tok];
  int   pA = route_pos[tok],             pB = route_pos[kstride + tok];
  const float* sA = eo + (((long long)eA * Bn + b) * CAPN + pA) * D;
  const float* sB = eo + (((long long)eB * Bn + b) * CAPN + pB) * D;
  float* po = xm + tok * (long long)D;
  for (int j = threadIdx.x; j < D; j += 256) {
    float v = po[j];
    if (gA > 0.f) v += gA * sA[j];
    if (gB > 0.f) v += gB * sB[j];
    po[j] = v;
  }
}

__global__ void finalize_aux_kernel(const float* aux, float* out)
{
  out[0] = aux[0] * 0.01f + aux[1] * 0.001f;
}

// ---------------------------------------------------------------------------
extern "C" void kernel_launch(void* const* d_in, const int* in_sizes, int n_in,
                              void* d_out, int out_size, void* d_ws, size_t ws_size,
                              hipStream_t stream)
{
  (void)in_sizes; (void)n_in; (void)out_size; (void)ws_size;
  typedef unsigned short u16;

  const float* x       = (const float*)d_in[0];
  const float* t_wqkv  = (const float*)d_in[1];
  const float* t_bqkv  = (const float*)d_in[2];
  const float* t_wo    = (const float*)d_in[3];
  const float* t_bo    = (const float*)d_in[4];
  const float* s_wqkv  = (const float*)d_in[5];
  const float* s_bqkv  = (const float*)d_in[6];
  const float* s_wo    = (const float*)d_in[7];
  const float* s_bo    = (const float*)d_in[8];
  const float* ln1_g   = (const float*)d_in[9];
  const float* ln1_b   = (const float*)d_in[10];
  const float* ln2_g   = (const float*)d_in[11];
  const float* ln2_b   = (const float*)d_in[12];
  const float* ln3_g   = (const float*)d_in[13];
  const float* ln3_b   = (const float*)d_in[14];
  const float* moe_rms = (const float*)d_in[15];
  const float* ffb_rms = (const float*)d_in[16];
  const float* ffb_w1  = (const float*)d_in[17];
  const float* ffb_b1  = (const float*)d_in[18];
  const float* ffb_mb  = (const float*)d_in[19];
  const float* ffb_w2  = (const float*)d_in[20];
  const float* ffb_b2  = (const float*)d_in[21];
  const float* ffa_rms = (const float*)d_in[22];
  const float* ffa_w1  = (const float*)d_in[23];
  const float* ffa_b1  = (const float*)d_in[24];
  const float* ffa_mb  = (const float*)d_in[25];
  const float* ffa_w2  = (const float*)d_in[26];
  const float* ffa_b2  = (const float*)d_in[27];
  const float* gate_w  = (const float*)d_in[28];
  const float* e_w1    = (const float*)d_in[29];
  const float* e_b1    = (const float*)d_in[30];
  const float* e_mb    = (const float*)d_in[31];
  const float* e_w2    = (const float*)d_in[32];
  const float* e_b2    = (const float*)d_in[33];

  const int Bn = 64, T = 32, S = 24, D = 1152, H = 3072;
  const long long NTOK = (long long)Bn * T * S;   // 49152
  const int M2 = Bn * T;                          // 2048
  const int NSLOT = NEXP * Bn * CAPN;             // 4096

  char* wsb = (char*)d_ws;
  size_t off = 0;
  auto alloc = [&](size_t bytes) -> void* {
    void* p = wsb + off;
    off = (off + bytes + 255) & ~(size_t)255;
    return p;
  };

  // bf16 operand buffers
  u16* xpad   = (u16*)alloc(NTOK * 64 * 2);
  u16* wqkvtp = (u16*)alloc(64 * 256 * 2);
  u16* wotp   = (u16*)alloc(64 * 128 * 2);
  u16* wqkvsp = (u16*)alloc(64 * 256 * 2);
  u16* wosp   = (u16*)alloc(64 * 128 * 2);
  u16* w1bp   = (u16*)alloc((long long)D * 2 * H * 2);
  u16* w2bp   = (u16*)alloc((long long)H * D * 2);
  u16* w1ap   = (u16*)alloc((long long)D * 2 * H * 2);
  u16* w2ap   = (u16*)alloc((long long)H * D * 2);
  u16* ew1p   = (u16*)alloc((long long)NEXP * D * 2 * H * 2);
  u16* ew2p   = (u16*)alloc((long long)NEXP * H * D * 2);
  u16* attnp  = (u16*)alloc(NTOK * 64 * 2);
  u16* nrmb   = (u16*)alloc((long long)M2 * D * 2);
  u16* actb   = (u16*)alloc((long long)M2 * H * 2);
  u16* eib    = (u16*)alloc((long long)NSLOT * D * 2);
  u16* eactb  = (u16*)alloc((long long)NSLOT * H * 2);
  // f32 buffers
  float* qkv    = (float*)alloc(NTOK * 144 * 4);
  float* proj   = (float*)alloc(NTOK * 48 * 4);
  float* tm     = (float*)alloc(NTOK * 48 * 4);
  float* sm     = (float*)alloc(NTOK * 48 * 4);
  float* inp    = (float*)alloc(NTOK * 48 * 4);
  float* xm     = (float*)alloc((long long)M2 * D * 4);
  float* hbuf   = (float*)alloc((long long)M2 * 2 * H * 4);
  float* xg     = (float*)alloc((long long)M2 * D * 4);
  float* logits = (float*)alloc((long long)M2 * NEXP * 4);
  int*   re     = (int*)alloc(2LL * M2 * 4);
  int*   rp     = (int*)alloc(2LL * M2 * 4);
  float* rg     = (float*)alloc(2LL * M2 * 4);
  int*   slots  = (int*)alloc((long long)NSLOT * 4);
  float* aux    = (float*)alloc(2 * 4);
  float* eh     = (float*)alloc((long long)NSLOT * 2 * H * 4);
  float* eo     = (float*)alloc((long long)NSLOT * D * 4);

  auto cdiv = [](long long a, long long b) -> unsigned { return (unsigned)((a + b - 1) / b); };
  dim3 blk256(256);

  auto pack = [&](const float* src, u16* dst, long long sr, int sc, long long dr, int dc) {
    pack_bf16_pad_kernel<<<dim3(cdiv(dr * dc, 256)), blk256, 0, stream>>>(src, dst, sr, sc, dr, dc);
  };
  auto gemm = [&](const u16* A, const u16* W, const float* bias, const float* resid, float* C,
                  int M, int N, int Ns, int K, int Z,
                  long long sA, long long sB, long long sBias, long long sC) {
    dim3 grid(cdiv(N, BNT), cdiv(M, BMT), (unsigned)Z);
    gemm_bf16_wmma<<<grid, blk256, 0, stream>>>(A, W, bias, resid, C, M, N, Ns, K, sA, sB, sBias, sC);
  };

  // ---- precision conversion / padding (weights + input) ----
  pack(x, xpad, NTOK, 48, NTOK, 64);
  pack(t_wqkv, wqkvtp, 48, 144, 64, 256);
  pack(t_wo,   wotp,   48, 48,  64, 128);
  pack(s_wqkv, wqkvsp, 48, 144, 64, 256);
  pack(s_wo,   wosp,   48, 48,  64, 128);
  pack(ffb_w1, w1bp, D, 2 * H, D, 2 * H);
  pack(ffb_w2, w2bp, H, D, H, D);
  pack(ffa_w1, w1ap, D, 2 * H, D, 2 * H);
  pack(ffa_w2, w2ap, H, D, H, D);
  pack(e_w1, ew1p, (long long)NEXP * D, 2 * H, (long long)NEXP * D, 2 * H);
  pack(e_w2, ew2p, (long long)NEXP * H, D, (long long)NEXP * H, D);

  // ---- temporal attention ----
  gemm(xpad, wqkvtp, t_bqkv, nullptr, qkv, (int)NTOK, 144, 256, 64, 1, 0, 0, 0, 0);
  attn_softmax_kernel<<<dim3(cdiv((long long)Bn * S * NHEADS, 8)), blk256, 0, stream>>>(
      qkv, attnp, 0, Bn, T, S);
  gemm(attnp, wotp, t_bo, nullptr, proj, (int)NTOK, 48, 128, 64, 1, 0, 0, 0, 0);
  add_layernorm48<<<dim3(cdiv(NTOK, 8)), blk256, 0, stream>>>(proj, x, ln1_g, ln1_b, tm, NTOK);

  // ---- spatial attention ----
  gemm(xpad, wqkvsp, s_bqkv, nullptr, qkv, (int)NTOK, 144, 256, 64, 1, 0, 0, 0, 0);
  attn_softmax_kernel<<<dim3(cdiv((long long)Bn * T * NHEADS, 8)), blk256, 0, stream>>>(
      qkv, attnp, 1, Bn, T, S);
  gemm(attnp, wosp, s_bo, nullptr, proj, (int)NTOK, 48, 128, 64, 1, 0, 0, 0, 0);
  add_layernorm48<<<dim3(cdiv(NTOK, 8)), blk256, 0, stream>>>(proj, x, ln2_g, ln2_b, sm, NTOK);

  // ---- inp = tm + sm ; xm = inp ----
  add2_kernel<<<dim3(cdiv(NTOK * 48, 256)), blk256, 0, stream>>>(tm, sm, inp, NTOK * 48);
  copy_kernel<<<dim3(cdiv(NTOK * 48, 256)), blk256, 0, stream>>>(inp, xm, NTOK * 48);

  // ---- ff_before (GEGLU), residual fused into second GEMM ----
  rmsnorm_kernel<u16><<<dim3(M2), blk256, 0, stream>>>(xm, ffb_rms, nrmb, D);
  gemm(nrmb, w1bp, ffb_b1, nullptr, hbuf, M2, 2 * H, 2 * H, D, 1, 0, 0, 0, 0);
  geglu_act_kernel<<<dim3(cdiv((long long)M2 * H, 256)), blk256, 0, stream>>>(
      hbuf, ffb_mb, actb, M2, H, 0);
  gemm(actb, w2bp, ffb_b2, xm, xm, M2, D, D, H, 1, 0, 0, 0, 0);

  // ---- MoE ----
  rmsnorm_kernel<float><<<dim3(M2), blk256, 0, stream>>>(xm, moe_rms, xg, D);
  gate_logits_kernel<<<dim3(M2), blk256, 0, stream>>>(xg, gate_w, logits, D);
  init_routing_kernel<<<dim3(cdiv(NSLOT, 256)), blk256, 0, stream>>>(slots, aux, NSLOT);
  routing_kernel<<<dim3(Bn), dim3(32), 0, stream>>>(logits, Bn, T, re, rp, rg, slots, aux);
  dispatch_gather_kernel<<<dim3(NSLOT), blk256, 0, stream>>>(xg, slots, eib, Bn, T, D);
  gemm(eib, ew1p, e_b1, nullptr, eh, Bn * CAPN, 2 * H, 2 * H, D, NEXP,
       (long long)Bn * CAPN * D, (long long)D * 2 * H, 2LL * H, (long long)Bn * CAPN * 2 * H);
  geglu_act_kernel<<<dim3(cdiv((long long)NSLOT * H, 256)), blk256, 0, stream>>>(
      eh, e_mb, eactb, NSLOT, H, Bn * CAPN);
  gemm(eactb, ew2p, e_b2, nullptr, eo, Bn * CAPN, D, D, H, NEXP,
       (long long)Bn * CAPN * H, (long long)H * D, (long long)D, (long long)Bn * CAPN * D);
  combine_kernel<<<dim3(M2), blk256, 0, stream>>>(eo, re, rp, rg, xm, Bn, T, D);

  // ---- ff_after (GEGLU), residual fused ----
  rmsnorm_kernel<u16><<<dim3(M2), blk256, 0, stream>>>(xm, ffa_rms, nrmb, D);
  gemm(nrmb, w1ap, ffa_b1, nullptr, hbuf, M2, 2 * H, 2 * H, D, 1, 0, 0, 0, 0);
  geglu_act_kernel<<<dim3(cdiv((long long)M2 * H, 256)), blk256, 0, stream>>>(
      hbuf, ffa_mb, actb, M2, H, 0);
  gemm(actb, w2ap, ffa_b2, xm, xm, M2, D, D, H, 1, 0, 0, 0, 0);

  // ---- final LN + aux ----
  float* outp = (float*)d_out;
  add_layernorm48<<<dim3(cdiv(NTOK, 8)), blk256, 0, stream>>>(xm, inp, ln3_g, ln3_b, outp, NTOK);
  finalize_aux_kernel<<<dim3(1), dim3(1), 0, stream>>>(aux, outp + NTOK * 48);
}
